// SimpleDecoder_35862976921933
// MI455X (gfx1250) — matrix operand order repaired and checked by reading, not compiled, not run on hardware
//
#include <hip/hip_runtime.h>
#include <math.h>

// ---------------------------------------------------------------------------
// Problem constants (SimpleDecoder reference): B=2, S=2048, H=1024, 16 heads
// ---------------------------------------------------------------------------
#define BATCH  2
#define SEQ    2048
#define HID    1024
#define FFN    4096
#define NHEADS 16
#define HDIM   64
#define MROWS  (BATCH * SEQ)   // 4096 token rows

typedef __attribute__((ext_vector_type(16))) __bf16 v16bf;
typedef __attribute__((ext_vector_type(8)))  __bf16 v8bf;
typedef __attribute__((ext_vector_type(8)))  float  v8f;
typedef __attribute__((ext_vector_type(4)))  unsigned int u32x4;
typedef __attribute__((ext_vector_type(8)))  int i32x8;
typedef __attribute__((ext_vector_type(4)))  int i32x4;

union FragA { v16bf v; v8bf h[2]; };

__device__ __forceinline__ float half16_max(float v) {
  #pragma unroll
  for (int m = 8; m >= 1; m >>= 1) v = fmaxf(v, __shfl_xor(v, m, 32));
  return v;
}
__device__ __forceinline__ float half16_sum(float v) {
  #pragma unroll
  for (int m = 8; m >= 1; m >>= 1) v += __shfl_xor(v, m, 32);
  return v;
}

// ---------------------------------------------------------------------------
// fp32 -> bf16 conversion (grid-stride)
// ---------------------------------------------------------------------------
__global__ void f32_to_bf16(const float* __restrict__ in, __bf16* __restrict__ out, size_t n) {
  size_t i = (size_t)blockIdx.x * blockDim.x + threadIdx.x;
  size_t stride = (size_t)gridDim.x * blockDim.x;
  for (; i < n; i += stride) out[i] = (__bf16)in[i];
}

// ---------------------------------------------------------------------------
// TDM: 2-D tile load Global -> LDS via Tensor Data Mover (CDNA5 ISA ch.8).
// D# group0: count=1, lds_addr, 57-bit global_addr, type=2 ("image").
// D# group1: data_size=1 (2 bytes), tensor_dim0/1, tile_dim0/1, dim0 stride.
// Groups 2/3 (and the trailing group on the 6-arg clang-23 builtin) are zero
// for a 2-D tensor. Tracked with TENSORcnt.
// ---------------------------------------------------------------------------
#define HAVE_TDM __has_builtin(__builtin_amdgcn_tensor_load_to_lds)

#if HAVE_TDM
__device__ __forceinline__ void tdm_load_2d(const __bf16* gptr, unsigned lds_byte_off,
                                            unsigned tile_x, unsigned tile_y,
                                            unsigned tensor_x, unsigned tensor_y,
                                            unsigned row_stride_elems)
{
  unsigned long long ga = (unsigned long long)(uintptr_t)gptr;
  unsigned glo = __builtin_amdgcn_readfirstlane((unsigned)(ga & 0xffffffffu));
  unsigned ghi = __builtin_amdgcn_readfirstlane((unsigned)(ga >> 32));
  unsigned lds = __builtin_amdgcn_readfirstlane(lds_byte_off);

  u32x4 g0;
  g0[0] = 1u;                                   // count=1 user descriptor
  g0[1] = lds;                                  // LDS byte address
  g0[2] = glo;                                  // global_addr[31:0]
  g0[3] = (ghi & 0x01ffffffu) | (2u << 30);     // global_addr[56:32] | type=2

  i32x8 g1;
  g1[0] = 0x00010000;                           // data_size=1 (2B); mask/flags=0
  g1[1] = (int)((tensor_x & 0xffffu) << 16);    // [63:48]=tensor_dim0 lo16
  g1[2] = (int)((tensor_x >> 16) | ((tensor_y & 0xffffu) << 16));
  g1[3] = (int)((tensor_y >> 16) | (tile_x << 16));   // [127:112]=tile_dim0
  g1[4] = (int)tile_y;                          // [143:128]=tile_dim1, tile_dim2=0
  g1[5] = (int)row_stride_elems;                // tensor_dim0_stride[31:0]
  g1[6] = 0;                                    // stride[47:32]=0, dim1_stride lo=0
  g1[7] = 0;
  i32x4 z4 = {0, 0, 0, 0};
  i32x8 z8 = {0, 0, 0, 0, 0, 0, 0, 0};
  // clang-23 / therock-10.0 lane: 6-argument form
  __builtin_amdgcn_tensor_load_to_lds(g0, g1, z4, z4, z8, 0);
}
#endif

// ---------------------------------------------------------------------------
// bf16 WMMA GEMM with TDM double-buffered LDS staging:
//   C[M,N] = A[M,K] @ W[N,K]^T + bias[N]
// Block = 256 threads (8 waves) -> 32(M) x 256(N) tile; K chunked by 64.
// ---------------------------------------------------------------------------
__global__ __launch_bounds__(256)
void gemm_bf16_wmma(const __bf16* __restrict__ A, const __bf16* __restrict__ W,
                    const float* __restrict__ bias,
                    float* __restrict__ Cf, __bf16* __restrict__ Cb,
                    int M, int N, int K, int relu)
{
  const int lane = threadIdx.x & 31;
  const int wave = threadIdx.x >> 5;
  const int lr   = lane & 15;
  const int hl   = (lane >> 4) & 1;
  const int nw   = wave & 3;          // N sub-strip
  const int mt   = wave >> 2;         // M sub-tile
  const int mBlk = blockIdx.x * 32;
  const int nBlk = blockIdx.y * 256;
  const int m0   = mBlk + mt * 16;

  __shared__ __attribute__((aligned(128))) __bf16 Alds[2][32 * 64];
  __shared__ __attribute__((aligned(128))) __bf16 Blds[2][256 * 64];

  v8f zero = {0.f,0.f,0.f,0.f,0.f,0.f,0.f,0.f};
  v8f acc[4];
  #pragma unroll
  for (int j = 0; j < 4; ++j) acc[j] = zero;

  const int nch = K >> 6;             // K chunks of 64

#if HAVE_TDM
  if (wave == 0) {
    tdm_load_2d(A + (size_t)mBlk * K, (unsigned)(uintptr_t)&Alds[0][0],
                64u, 32u, (unsigned)K, (unsigned)M, (unsigned)K);
    tdm_load_2d(W + (size_t)nBlk * K, (unsigned)(uintptr_t)&Blds[0][0],
                64u, 256u, (unsigned)K, (unsigned)N, (unsigned)K);
  }
#endif

  for (int c = 0; c < nch; ++c) {
    const int buf = c & 1;
#if HAVE_TDM
    if (wave == 0) {
      if (c + 1 < nch) {
        tdm_load_2d(A + (size_t)mBlk * K + (c + 1) * 64, (unsigned)(uintptr_t)&Alds[buf ^ 1][0],
                    64u, 32u, (unsigned)K, (unsigned)M, (unsigned)K);
        tdm_load_2d(W + (size_t)nBlk * K + (c + 1) * 64, (unsigned)(uintptr_t)&Blds[buf ^ 1][0],
                    64u, 256u, (unsigned)K, (unsigned)N, (unsigned)K);
        __builtin_amdgcn_s_wait_tensorcnt(2);   // chunk c's two loads complete
      } else {
        __builtin_amdgcn_s_wait_tensorcnt(0);
      }
    }
#else
    // Fallback: cooperative copy Global -> LDS (all 256 threads)
    {
      const __bf16* ag = A + (size_t)mBlk * K + c * 64;
      for (int r = threadIdx.x / 8; r < 32; r += 32) {
        int cc = (threadIdx.x & 7) * 8;
        *(v8bf*)&Alds[buf][r * 64 + cc] = *(const v8bf*)(ag + (size_t)r * K + cc);
      }
      const __bf16* wg = W + (size_t)nBlk * K + c * 64;
      for (int r = threadIdx.x / 8; r < 256; r += 32) {
        int cc = (threadIdx.x & 7) * 8;
        *(v8bf*)&Blds[buf][r * 64 + cc] = *(const v8bf*)(wg + (size_t)r * K + cc);
      }
    }
#endif
    __syncthreads();                  // publish chunk c LDS data to all waves

    const __bf16* Ab = &Alds[buf][0];
    const __bf16* Bb = &Blds[buf][0];
    #pragma unroll
    for (int ks = 0; ks < 2; ++ks) {  // two K=32 WMMA steps per chunk
      FragA af;
      const __bf16* ap = Ab + (mt * 16 + lr) * 64 + ks * 32 + (hl ? 16 : 0);
      af.h[0] = *(const v8bf*)(ap);
      af.h[1] = *(const v8bf*)(ap + 8);
      #pragma unroll
      for (int j = 0; j < 4; ++j) {
        const __bf16* bp = Bb + (nw * 64 + j * 16 + lr) * 64 + ks * 32 + (hl ? 16 : 0);
        v16bf bfrag = *(const v16bf*)bp;
        acc[j] = __builtin_amdgcn_wmma_f32_16x16x32_bf16(
            false, af.v, false, bfrag, (short)0, acc[j], false, false);
      }
    }
    __syncthreads();                  // compute done before buffer reuse
  }

  // C/D layout: lane 0-15 -> N=lr, M=r; lane 16-31 -> N=lr, M=r+8
  #pragma unroll
  for (int j = 0; j < 4; ++j) {
    int n = nBlk + nw * 64 + j * 16 + lr;
    float bv = bias[n];
    #pragma unroll
    for (int r = 0; r < 8; ++r) {
      float v = acc[j][r] + bv;
      if (relu) v = fmaxf(v, 0.f);
      size_t idx = (size_t)(m0 + r + hl * 8) * N + n;
      if (Cf) Cf[idx] = v;
      if (Cb) Cb[idx] = (__bf16)v;
    }
  }
}

// ---------------------------------------------------------------------------
// Causal flash attention, one wave per (batch, head, 16-query tile).
// ---------------------------------------------------------------------------
__global__ __launch_bounds__(32)
void attn_fwd(const __bf16* __restrict__ Q, const __bf16* __restrict__ Kmat,
              const __bf16* __restrict__ Vmat, __bf16* __restrict__ O)
{
  const int lane = threadIdx.x & 31;
  const int lr   = lane & 15;
  const int hl   = (lane & 16) ? 1 : 0;
  const int qt   = blockIdx.x;
  const int b    = blockIdx.y / NHEADS;
  const int h    = blockIdx.y % NHEADS;
  const int c0   = h * HDIM;
  const size_t rowbase = (size_t)b * SEQ;

  __shared__ __attribute__((aligned(32))) __bf16 Plds[16 * 32];
  __shared__ __attribute__((aligned(32))) __bf16 Vlds[HDIM * 32];

  FragA qf[2];
  {
    const __bf16* qrow = Q + (rowbase + qt * 16 + lr) * HID + c0 + (hl ? 16 : 0);
    qf[0].h[0] = *(const v8bf*)(qrow);
    qf[0].h[1] = *(const v8bf*)(qrow + 8);
    qf[1].h[0] = *(const v8bf*)(qrow + 32);
    qf[1].h[1] = *(const v8bf*)(qrow + 40);
  }

  v8f zero = {0.f,0.f,0.f,0.f,0.f,0.f,0.f,0.f};
  v8f o[4];
  #pragma unroll
  for (int j = 0; j < 4; ++j) o[j] = zero;
  float mrow[8], lrow[8];
  #pragma unroll
  for (int r = 0; r < 8; ++r) { mrow[r] = -1e30f; lrow[r] = 0.f; }

  const int   kend  = ((qt + 1) * 16 + 31) & ~31;
  const float scale = 0.125f;

  for (int kb = 0; kb < kend; kb += 32) {
    {
      const __bf16* vrow = Vmat + (rowbase + kb + lane) * HID + c0;
      #pragma unroll
      for (int c = 0; c < 8; ++c) {
        v8bf t = *(const v8bf*)(vrow + c * 8);
        #pragma unroll
        for (int e = 0; e < 8; ++e) Vlds[(c * 8 + e) * 32 + lane] = t[e];
      }
    }

    v8f sc[2];
    #pragma unroll
    for (int kt = 0; kt < 2; ++kt) {
      v8f s = zero;
      #pragma unroll
      for (int c = 0; c < 2; ++c) {
        const __bf16* kp = Kmat + (rowbase + kb + kt * 16 + lr) * HID
                           + c0 + c * 32 + (hl ? 16 : 0);
        v16bf kfrag = *(const v16bf*)kp;
        s = __builtin_amdgcn_wmma_f32_16x16x32_bf16(
            false, qf[c].v, false, kfrag, (short)0, s, false, false);
      }
      sc[kt] = s;
    }

    #pragma unroll
    for (int r = 0; r < 8; ++r) {
      int row = qt * 16 + r + hl * 8;
      float v0 = sc[0][r] * scale;
      float v1 = sc[1][r] * scale;
      v0 = (kb + lr      > row) ? -1e30f : v0;
      v1 = (kb + 16 + lr > row) ? -1e30f : v1;
      float mx   = half16_max(fmaxf(v0, v1));
      float mnew = fmaxf(mrow[r], mx);
      float corr = __expf(mrow[r] - mnew);
      float p0   = __expf(v0 - mnew);
      float p1   = __expf(v1 - mnew);
      float srow = half16_sum(p0 + p1);
      lrow[r] = lrow[r] * corr + srow;
      mrow[r] = mnew;
      #pragma unroll
      for (int j = 0; j < 4; ++j) o[j][r] *= corr;
      Plds[(r + hl * 8) * 32 + lr]      = (__bf16)p0;
      Plds[(r + hl * 8) * 32 + 16 + lr] = (__bf16)p1;
    }
    __syncthreads();

    FragA pf;
    {
      const __bf16* pp = Plds + lr * 32 + (hl ? 16 : 0);
      pf.h[0] = *(const v8bf*)(pp);
      pf.h[1] = *(const v8bf*)(pp + 8);
    }
    #pragma unroll
    for (int j = 0; j < 4; ++j) {
      const __bf16* vp = Vlds + (j * 16 + lr) * 32 + (hl ? 16 : 0);
      v16bf vfrag = *(const v16bf*)vp;
      o[j] = __builtin_amdgcn_wmma_f32_16x16x32_bf16(
          false, pf.v, false, vfrag, (short)0, o[j], false, false);
    }
    __syncthreads();
  }

  #pragma unroll
  for (int j = 0; j < 4; ++j) {
    #pragma unroll
    for (int r = 0; r < 8; ++r) {
      float val = o[j][r] / lrow[r];
      size_t idx = (rowbase + qt * 16 + r + hl * 8) * HID + c0 + j * 16 + lr;
      O[idx] = (__bf16)val;
    }
  }
}

// ---------------------------------------------------------------------------
// Fused residual + LayerNorm over H=1024.
// ---------------------------------------------------------------------------
__global__ __launch_bounds__(256)
void ln_residual(const float* __restrict__ A, const float* __restrict__ Badd,
                 const float* __restrict__ g, const float* __restrict__ bt,
                 float* __restrict__ outF, __bf16* __restrict__ outB)
{
  const int row = blockIdx.x;
  const float* a  = A    + (size_t)row * HID;
  const float* bb = Badd + (size_t)row * HID;

  float vals[4], lsum = 0.f, lsq = 0.f;
  #pragma unroll
  for (int i = 0; i < 4; ++i) {
    int c = threadIdx.x + i * 256;
    float v = a[c] + bb[c];
    vals[i] = v; lsum += v; lsq += v * v;
  }
  __shared__ float s1[256], s2[256];
  s1[threadIdx.x] = lsum; s2[threadIdx.x] = lsq;
  __syncthreads();
  for (int st = 128; st > 0; st >>= 1) {
    if (threadIdx.x < st) {
      s1[threadIdx.x] += s1[threadIdx.x + st];
      s2[threadIdx.x] += s2[threadIdx.x + st];
    }
    __syncthreads();
  }
  float mu  = s1[0] * (1.f / HID);
  float var = s2[0] * (1.f / HID) - mu * mu;
  float rs  = rsqrtf(var + 1e-5f);
  #pragma unroll
  for (int i = 0; i < 4; ++i) {
    int c = threadIdx.x + i * 256;
    float ov = (vals[i] - mu) * rs * g[c] + bt[c];
    size_t idx = (size_t)row * HID + c;
    if (outF) outF[idx] = ov;
    if (outB) outB[idx] = (__bf16)ov;
  }
}

// ---------------------------------------------------------------------------
// Host-side orchestration
// ---------------------------------------------------------------------------
extern "C" void kernel_launch(void* const* d_in, const int* in_sizes, int n_in,
                              void* d_out, int out_size, void* d_ws, size_t ws_size,
                              hipStream_t stream) {
  (void)in_sizes; (void)n_in; (void)out_size; (void)ws_size;

  const float* X     = (const float*)d_in[0];
  const float* Wq    = (const float*)d_in[1];
  const float* bq    = (const float*)d_in[2];
  const float* Wk    = (const float*)d_in[3];
  const float* bk    = (const float*)d_in[4];
  const float* Wv    = (const float*)d_in[5];
  const float* bv    = (const float*)d_in[6];
  const float* Wo    = (const float*)d_in[7];
  const float* bo    = (const float*)d_in[8];
  const float* ln1g  = (const float*)d_in[9];
  const float* ln1b  = (const float*)d_in[10];
  const float* Wup   = (const float*)d_in[11];
  const float* bup   = (const float*)d_in[12];
  const float* Wdown = (const float*)d_in[13];
  const float* bdown = (const float*)d_in[14];
  const float* ln2g  = (const float*)d_in[15];
  const float* ln2b  = (const float*)d_in[16];

  char* ws = (char*)d_ws;
  size_t off = 0;
  auto alloc = [&](size_t bytes) -> void* {
    void* p = ws + off;
    off += (bytes + 255) & ~(size_t)255;
    return p;
  };

  const size_t M = MROWS;
  __bf16* Xb   = (__bf16*)alloc(M * HID * 2);
  __bf16* Wqb  = (__bf16*)alloc((size_t)HID * HID * 2);
  __bf16* Wkb  = (__bf16*)alloc((size_t)HID * HID * 2);
  __bf16* Wvb  = (__bf16*)alloc((size_t)HID * HID * 2);
  __bf16* Wob  = (__bf16*)alloc((size_t)HID * HID * 2);
  __bf16* Wupb = (__bf16*)alloc((size_t)FFN * HID * 2);
  __bf16* Wdnb = (__bf16*)alloc((size_t)HID * FFN * 2);
  __bf16* Qb   = (__bf16*)alloc(M * HID * 2);
  __bf16* Kb   = (__bf16*)alloc(M * HID * 2);
  __bf16* Vb   = (__bf16*)alloc(M * HID * 2);
  __bf16* Attb = (__bf16*)alloc(M * HID * 2);
  float*  AttF = (float*)alloc(M * HID * 4);
  float*  Hf   = (float*)alloc(M * HID * 4);
  __bf16* Hb   = (__bf16*)alloc(M * HID * 2);
  __bf16* UpB  = (__bf16*)alloc(M * FFN * 2);
  float*  DnF  = (float*)alloc(M * HID * 4);

  // 1) precision conversion
  f32_to_bf16<<<1024, 256, 0, stream>>>(X,     Xb,   M * (size_t)HID);
  f32_to_bf16<<<1024, 256, 0, stream>>>(Wq,    Wqb,  (size_t)HID * HID);
  f32_to_bf16<<<1024, 256, 0, stream>>>(Wk,    Wkb,  (size_t)HID * HID);
  f32_to_bf16<<<1024, 256, 0, stream>>>(Wv,    Wvb,  (size_t)HID * HID);
  f32_to_bf16<<<1024, 256, 0, stream>>>(Wo,    Wob,  (size_t)HID * HID);
  f32_to_bf16<<<1024, 256, 0, stream>>>(Wup,   Wupb, (size_t)FFN * HID);
  f32_to_bf16<<<1024, 256, 0, stream>>>(Wdown, Wdnb, (size_t)HID * FFN);

  // 2) QKV projections (block tile 32x256)
  dim3 gH(MROWS / 32, HID / 256);
  gemm_bf16_wmma<<<gH, 256, 0, stream>>>(Xb, Wqb, bq, nullptr, Qb, MROWS, HID, HID, 0);
  gemm_bf16_wmma<<<gH, 256, 0, stream>>>(Xb, Wkb, bk, nullptr, Kb, MROWS, HID, HID, 0);
  gemm_bf16_wmma<<<gH, 256, 0, stream>>>(Xb, Wvb, bv, nullptr, Vb, MROWS, HID, HID, 0);

  // 3) causal flash attention
  dim3 gA(SEQ / 16, BATCH * NHEADS);
  attn_fwd<<<gA, 32, 0, stream>>>(Qb, Kb, Vb, Attb);

  // 4) output projection + residual + LN1
  gemm_bf16_wmma<<<gH, 256, 0, stream>>>(Attb, Wob, bo, AttF, nullptr, MROWS, HID, HID, 0);
  ln_residual<<<MROWS, 256, 0, stream>>>(X, AttF, ln1g, ln1b, Hf, Hb);

  // 5) MLP up (ReLU fused) -> down -> residual + LN2 into d_out
  dim3 gF(MROWS / 32, FFN / 256);
  gemm_bf16_wmma<<<gF, 256, 0, stream>>>(Hb, Wupb, bup, nullptr, UpB, MROWS, FFN, HID, 1);
  gemm_bf16_wmma<<<gH, 256, 0, stream>>>(UpB, Wdnb, bdown, DnF, nullptr, MROWS, HID, FFN, 0);
  ln_residual<<<MROWS, 256, 0, stream>>>(Hf, DnF, ln2g, ln2b, (float*)d_out, nullptr);
}
